// Adaptive_Softmax_11879879541904
// MI455X (gfx1250) — matrix operand order repaired and checked by reading, not compiled, not run on hardware
//
#include <hip/hip_runtime.h>
#include <hip/hip_bf16.h>
#include <math.h>

// ---------------------------------------------------------------------------
// Adaptive softmax NLL for MI455X (gfx1250, wave32, WMMA).
// Fused GEMM(4096x50000x512) -> per-row segmented logsumexp using split-bf16
// (3x v_wmma_f32_16x16x32_bf16 per K=32 chunk, ~fp32 accuracy at bf16 rate).
// Each wave computes 16 rows x 32 cols per K sweep so the x-row LDS fragments
// are amortized over two column tiles (2.0 ds_load_b128 per WMMA).
// ---------------------------------------------------------------------------

typedef __attribute__((ext_vector_type(16))) __bf16 v16bf;
typedef __attribute__((ext_vector_type(8)))  float  v8f;

#define HID     512
#define VOCABSZ 50000
#define NROWS   4096
#define M_TILE  64             // rows per block (4 row-groups x 16)
#define NCOL    64             // columns staged per epoch (2 col-groups x 32)
#define NSPLIT  4              // N splits per segment (grid.y)
#define PPARTS  (NSPLIT * 2)   // partials per (row, seg): NSPLIT x 2 col-groups
#define KPAD    520            // 512+8 bf16 pad: 1040B stride, 16B aligned, bank-spread
#define NEGINF  (-__builtin_inff())

__device__ __forceinline__ void split_bf16(float f, __bf16& h, __bf16& l) {
  h = (__bf16)f;
  l = (__bf16)(f - (float)h);
}

// ---------------------------------------------------------------------------
// Kernel 1: online (max, sumexp) per (row, segment, nsplit, colgroup) over
// z = x . W[:, cols] + b.  Partials go to ws as float2.
// ---------------------------------------------------------------------------
__global__ __launch_bounds__(256)
void asoft_gemm_lse(const float* __restrict__ x,
                    const float* __restrict__ W,    // (HID, VOCAB) row-major
                    const float* __restrict__ Wb,   // (VOCAB)
                    float2* __restrict__ ws_ms)     // [row][seg][PPARTS]
{
  __shared__ __align__(16) __bf16 sAh[M_TILE * KPAD];  // x rows, hi
  __shared__ __align__(16) __bf16 sAl[M_TILE * KPAD];  // x rows, lo
  __shared__ __align__(16) __bf16 sBh[NCOL * KPAD];    // W cols, hi
  __shared__ __align__(16) __bf16 sBl[NCOL * KPAD];    // W cols, lo

  const int t     = threadIdx.x;
  const int mBase = blockIdx.x * M_TILE;
  const int ys    = blockIdx.y;

  // ---- stage x rows (64 x 512 f32) as split bf16, once ----
  #pragma unroll 4
  for (int i = 0; i < 32; ++i) {
    int idx4 = t + i * 256;              // [0, 8192) float4s
    int row  = idx4 >> 7;                // 128 float4 per row
    int k    = (idx4 & 127) * 4;
    int rG   = mBase + row;
    int bb = rG >> 10, ll = rG & 1023;   // x is (4, 1025, 512); drop last token
    const float4 v = *(const float4*)&x[((size_t)(bb * 1025 + ll)) * HID + k];
    float arr[4] = {v.x, v.y, v.z, v.w};
    #pragma unroll
    for (int j = 0; j < 4; ++j) {
      __bf16 h, l;
      split_bf16(arr[j], h, l);
      sAh[row * KPAD + k + j] = h;
      sAl[row * KPAD + k + j] = l;
    }
  }
  __syncthreads();

  const int lane = t & 31;
  const int wv   = t >> 5;
  const int rg   = wv >> 1;              // row group 0..3
  const int cg   = wv & 1;               // col group 0..1 (32 cols each)
  const int lm   = lane & 15;
  const int half = lane >> 4;
  const int rowLoc = rg * 16 + lm;       // this lane's x-row (D's N index)
  const int cut[4] = {0, 2000, 10000, VOCABSZ};

  for (int seg = 0; seg < 3; ++seg) {
    const int lo = cut[seg];
    const int w  = cut[seg + 1] - lo;
    const int nS = lo + (int)(((long long)w * ys       / NSPLIT) & ~15LL);
    const int nE = lo + (int)(((long long)w * (ys + 1) / NSPLIT) & ~15LL);

    float mr = NEGINF;   // running max    (per lane = per row, this col group)
    float sr = 0.f;      // running sumexp

    for (int nb = nS; nb < nE; nb += NCOL) {
      __syncthreads();   // previous epoch's compute done
      // ---- stage W epoch (512 x 64 f32) as split bf16, [col][k]; clamp OOB ----
      #pragma unroll 4
      for (int i = 0; i < 32; ++i) {
        int idx4 = t + i * 256;          // [0, 8192)
        int k    = idx4 >> 4;            // 16 float4 per k-row
        int c4   = (idx4 & 15) * 4;
        int cG   = nb + c4;
        if (cG > VOCABSZ - 4) cG = VOCABSZ - 4;   // stay in-bounds; masked later
        const float4 v = *(const float4*)&W[(size_t)k * VOCABSZ + cG];
        float arr[4] = {v.x, v.y, v.z, v.w};
        #pragma unroll
        for (int j = 0; j < 4; ++j) {
          __bf16 h, l;
          split_bf16(arr[j], h, l);
          sBh[(c4 + j) * KPAD + k] = h;
          sBl[(c4 + j) * KPAD + k] = l;
        }
      }
      __syncthreads();

      // ---- two 16x16 tiles: D[M=16 W-cols][N=16 x-rows], K=512 ----
      v8f acc0 = {0.f, 0.f, 0.f, 0.f, 0.f, 0.f, 0.f, 0.f};
      v8f acc1 = {0.f, 0.f, 0.f, 0.f, 0.f, 0.f, 0.f, 0.f};
      #pragma unroll
      for (int kc = 0; kc < HID; kc += 32) {
        union Frag { uint4 q[2]; v16bf v; };
        Frag xh, xl, wh0, wl0, wh1, wl1;
        // B operand (x rows): lane lm = row; K contiguous 16 per half
        {
          int bi = rowLoc * KPAD + kc + half * 16;
          xh.q[0] = *(const uint4*)&sAh[bi];
          xh.q[1] = *(const uint4*)&sAh[bi + 8];
          xl.q[0] = *(const uint4*)&sAl[bi];
          xl.q[1] = *(const uint4*)&sAl[bi + 8];
        }
        // A operand (W cols): lane lm = col; K contiguous 16 per half
        {
          int a0 = (cg * 32 + lm) * KPAD + kc + half * 16;
          wh0.q[0] = *(const uint4*)&sBh[a0];
          wh0.q[1] = *(const uint4*)&sBh[a0 + 8];
          wl0.q[0] = *(const uint4*)&sBl[a0];
          wl0.q[1] = *(const uint4*)&sBl[a0 + 8];
          int a1 = a0 + 16 * KPAD;
          wh1.q[0] = *(const uint4*)&sBh[a1];
          wh1.q[1] = *(const uint4*)&sBh[a1 + 8];
          wl1.q[0] = *(const uint4*)&sBl[a1];
          wl1.q[1] = *(const uint4*)&sBl[a1 + 8];
        }
        // split-bf16: hh + hl + lh  ~ fp32 product accuracy
        acc0 = __builtin_amdgcn_wmma_f32_16x16x32_bf16(false, wh0.v, false, xh.v,
                                                       (short)0, acc0, false, false);
        acc1 = __builtin_amdgcn_wmma_f32_16x16x32_bf16(false, wh1.v, false, xh.v,
                                                       (short)0, acc1, false, false);
        acc0 = __builtin_amdgcn_wmma_f32_16x16x32_bf16(false, wh0.v, false, xl.v,
                                                       (short)0, acc0, false, false);
        acc1 = __builtin_amdgcn_wmma_f32_16x16x32_bf16(false, wh1.v, false, xl.v,
                                                       (short)0, acc1, false, false);
        acc0 = __builtin_amdgcn_wmma_f32_16x16x32_bf16(false, wl0.v, false, xh.v,
                                                       (short)0, acc0, false, false);
        acc1 = __builtin_amdgcn_wmma_f32_16x16x32_bf16(false, wl1.v, false, xh.v,
                                                       (short)0, acc1, false, false);
      }

      // ---- prefetch next W epoch into WGP$ (global_prefetch_b8) ----
      if (nb + NCOL < nE) {
        const float* pf = &W[(size_t)t * VOCABSZ + nb + NCOL];
        __builtin_prefetch(pf, 0, 1);
        __builtin_prefetch(pf + (size_t)256 * VOCABSZ, 0, 1);
      }

      // ---- online logsumexp: lane holds 8 cols of ONE row per tile ----
      #pragma unroll
      for (int tt = 0; tt < 2; ++tt) {
        v8f acc = tt ? acc1 : acc0;
        int c0  = nb + cg * 32 + tt * 16 + half * 8;   // first col of this slot run
        int cb0 = (c0 > VOCABSZ - 8) ? (VOCABSZ - 8) : c0;
        float4 b0 = *(const float4*)&Wb[cb0];
        float4 b1 = *(const float4*)&Wb[cb0 + 4];
        float bias[8] = {b0.x, b0.y, b0.z, b0.w, b1.x, b1.y, b1.z, b1.w};
        float z[8], tm = NEGINF;
        #pragma unroll
        for (int v = 0; v < 8; ++v) {
          bool ok = (c0 + v) < nE;                     // mask tail / OOB columns
          z[v] = ok ? (acc[v] + bias[v]) : NEGINF;
          tm = fmaxf(tm, z[v]);
        }
        float te = 0.f;
        #pragma unroll
        for (int v = 0; v < 8; ++v)
          te += (z[v] > NEGINF) ? __expf(z[v] - tm) : 0.f;
        // merge the two 8-col halves (lane <-> lane^16, same row)
        float om = __shfl_xor(tm, 16);
        float oe = __shfl_xor(te, 16);
        float hm = fmaxf(tm, om);
        float he = 0.f;
        if (tm > NEGINF) he += te * __expf(tm - hm);
        if (om > NEGINF) he += oe * __expf(om - hm);
        // merge into running state
        float nm = fmaxf(mr, hm);
        float s2 = 0.f;
        if (mr > NEGINF) s2 += sr * __expf(mr - nm);
        if (hm > NEGINF) s2 += he * __expf(hm - nm);
        mr = nm; sr = s2;
      }
    }

    // ---- write partial (m, s): lanes 0..15 each own one row ----
    if (half == 0) {
      int rG = mBase + rowLoc;
      ws_ms[((size_t)rG * 3 + seg) * PPARTS + ys * 2 + cg] = make_float2(mr, sr);
    }
    __syncthreads();
  }
}

// ---------------------------------------------------------------------------
// Kernel 2: per row -- merge partials, cluster head, target logit, nll.
// One wave per row.
// ---------------------------------------------------------------------------
__global__ __launch_bounds__(256)
void asoft_combine(const float* __restrict__ x,
                   const int*   __restrict__ y,
                   const float* __restrict__ cw,   // (HID, 3)
                   const float* __restrict__ cb,   // (1, 3)
                   const float* __restrict__ W,    // (HID, VOCAB)
                   const float* __restrict__ Wb,   // (VOCAB)
                   const float2* __restrict__ ws_ms,
                   float* __restrict__ out)
{
  const int r = blockIdx.x * 8 + (threadIdx.x >> 5);
  if (r >= NROWS) return;
  const int lane = threadIdx.x & 31;
  const int bb = r >> 10, ll = r & 1023;
  const float* xr = &x[((size_t)(bb * 1025 + ll)) * HID];
  const int col = y[r];

  float c0 = 0.f, c1 = 0.f, c2 = 0.f, zt = 0.f;
  const int h0 = lane * 16;
  #pragma unroll
  for (int i = 0; i < 4; ++i) {
    float4 v = *(const float4*)&xr[h0 + i * 4];
    float arr[4] = {v.x, v.y, v.z, v.w};
    #pragma unroll
    for (int j = 0; j < 4; ++j) {
      int h = h0 + i * 4 + j;
      float xv = arr[j];
      c0 += xv * cw[h * 3 + 0];
      c1 += xv * cw[h * 3 + 1];
      c2 += xv * cw[h * 3 + 2];
      zt += xv * W[(size_t)h * VOCABSZ + col];
    }
  }
  #pragma unroll
  for (int m = 16; m >= 1; m >>= 1) {
    c0 += __shfl_xor(c0, m);
    c1 += __shfl_xor(c1, m);
    c2 += __shfl_xor(c2, m);
    zt += __shfl_xor(zt, m);
  }

  if (lane == 0) {
    c0 += cb[0]; c1 += cb[1]; c2 += cb[2];
    zt += Wb[col];
    const int seg = (col < 2000) ? 0 : ((col < 10000) ? 1 : 2);
    // cluster log-softmax, select this row's cluster
    float cm = fmaxf(c0, fmaxf(c1, c2));
    float cs = __expf(c0 - cm) + __expf(c1 - cm) + __expf(c2 - cm);
    float csel = (seg == 0) ? c0 : ((seg == 1) ? c1 : c2);
    float cll = (csel - cm) - __logf(cs);
    // merge the PPARTS (m, s) partials of this segment
    float M = NEGINF, S = 0.f;
    const float2* p = &ws_ms[((size_t)r * 3 + seg) * PPARTS];
    for (int ysp = 0; ysp < PPARTS; ++ysp) {
      float2 ms = p[ysp];
      if (!(ms.x > NEGINF)) continue;
      float nm = fmaxf(M, ms.x);
      float s2 = ms.y * __expf(ms.x - nm);
      if (M > NEGINF) s2 += S * __expf(M - nm);
      S = s2; M = nm;
    }
    float lse = M + __logf(S);
    out[r] = -(cll + zt - lse);
  }
}

// ---------------------------------------------------------------------------
extern "C" void kernel_launch(void* const* d_in, const int* in_sizes, int n_in,
                              void* d_out, int out_size, void* d_ws, size_t ws_size,
                              hipStream_t stream) {
  const float* x  = (const float*)d_in[0];   // (4, 1025, 512) f32
  const int*   y  = (const int*)  d_in[1];   // (4, 1024) int
  const float* cw = (const float*)d_in[2];   // (512, 3)
  const float* cb = (const float*)d_in[3];   // (1, 3)
  const float* W  = (const float*)d_in[4];   // (512, 50000)
  const float* Wb = (const float*)d_in[5];   // (1, 50000)
  float*  out = (float*)d_out;               // (4096,)
  float2* ws  = (float2*)d_ws;               // 4096*3*PPARTS float2 = 768 KB

  dim3 g1(NROWS / M_TILE, NSPLIT);           // 64 x 4 = 256 blocks
  asoft_gemm_lse<<<g1, 256, 0, stream>>>(x, W, Wb, ws);
  asoft_combine<<<NROWS / 8, 256, 0, stream>>>(x, y, cw, cb, W, Wb, ws, out);
}
